// MultiViewMamba_87943750353310
// MI455X (gfx1250) — compile-verified, matrix operand order checked by reading, not run on hardware
//
#include <hip/hip_runtime.h>
#include <hip/hip_fp16.h>

typedef __attribute__((ext_vector_type(16))) _Float16 v16h;
typedef __attribute__((ext_vector_type(8)))  _Float16 v8h;
typedef __attribute__((ext_vector_type(8)))  float    v8f;

#define D_MODEL 256
#define D_STATE 16
#define N_VIEWS 4
#define D_CONV  4
#define D_INNER 512
#define DT_RANK 16
#define BB 8
#define LL 2048
#define MM (BB * LL)          // 16384 tokens per view
#define N_DBL (DT_RANK + 2 * D_STATE)   // 48

// ---------------------------------------------------------------------------
// f32 -> f16 conversion (weights)
// ---------------------------------------------------------------------------
__global__ void f32_to_f16_kernel(const float* __restrict__ in,
                                  _Float16* __restrict__ out, int n) {
    int i = blockIdx.x * blockDim.x + threadIdx.x;
    if (i < n) out[i] = (_Float16)in[i];
}

// ---------------------------------------------------------------------------
// LayerNorm over D_MODEL, one wave32 per token, output f16
// x layout: (B, L, V, D_MODEL); token m = b*L + l -> row ((m*V)+v)*256
// ---------------------------------------------------------------------------
__global__ void layernorm_f16_kernel(const float* __restrict__ x,
                                     const float* __restrict__ gma,
                                     const float* __restrict__ bta,
                                     _Float16* __restrict__ xn16, int v) {
    int wave = (blockIdx.x * blockDim.x + threadIdx.x) >> 5;
    int lane = threadIdx.x & 31;
    if (wave >= MM) return;
    const float* row = x + ((size_t)wave * N_VIEWS + v) * D_MODEL;
    float vals[8];
    float s = 0.0f, ss = 0.0f;
#pragma unroll
    for (int i = 0; i < 8; i++) {
        float t = row[lane + 32 * i];
        vals[i] = t; s += t; ss += t * t;
    }
#pragma unroll
    for (int off = 16; off >= 1; off >>= 1) {
        s  += __shfl_xor(s,  off, 32);
        ss += __shfl_xor(ss, off, 32);
    }
    float mu   = s * (1.0f / D_MODEL);
    float var  = ss * (1.0f / D_MODEL) - mu * mu;
    float rstd = rsqrtf(var + 1e-5f);
    _Float16* orow = xn16 + (size_t)wave * D_MODEL;
#pragma unroll
    for (int i = 0; i < 8; i++) {
        int d = lane + 32 * i;
        orow[d] = (_Float16)((vals[i] - mu) * rstd * gma[d] + bta[d]);
    }
}

// ---------------------------------------------------------------------------
// WMMA GEMM, fully compile-time tile shape (no divergent guards -> no spills):
//   Out[M, N](f32, row stride OUT_STRIDE) = X[M, K] @ W[N, K]^T
// One wave computes a (MT*16)(M) x (NT*16)(N) strip: MT*NT accumulators.
// All fragment loads for a k-step are issued as one clause before the WMMA
// group so the XDL pipe runs uninterrupted and overlaps the next loads.
// Fragment layout per CDNA5 ISA 7.12.2 (16-bit A 16x32 / B 32x16):
//   lane l: row (l & 15), g = l >> 4; halves 0..7 -> K = kb+8g.., halves 8..15
//   -> K = kb+16+8g.. ; C/D: elem (M = r + 8g, N = l & 15).
// ---------------------------------------------------------------------------
template <int N, int K, int NT, int MT, int OUT_STRIDE>
__global__ __launch_bounds__(256) void wmma_gemm_f16_kernel(
    const _Float16* __restrict__ X,
    const _Float16* __restrict__ W,
    float* __restrict__ Out, int M) {
    constexpr int TILES_N_GRP = (N + NT * 16 - 1) / (NT * 16);
    int wave = (blockIdx.x * blockDim.x + threadIdx.x) >> 5;
    int lane = threadIdx.x & 31;
    int tmg = wave / TILES_N_GRP;
    int tng = wave % TILES_N_GRP;
    if (tmg * MT * 16 >= M) return;
    int g = lane >> 4;
    int r = lane & 15;
    const _Float16* arow[MT];
#pragma unroll
    for (int mi = 0; mi < MT; mi++)
        arow[mi] = X + (size_t)(tmg * MT * 16 + mi * 16 + r) * K + 8 * g;
    const _Float16* brow[NT];
#pragma unroll
    for (int j = 0; j < NT; j++)
        brow[j] = W + (size_t)(tng * NT * 16 + j * 16 + r) * K + 8 * g;

    v8f acc[MT][NT] = {};
    for (int kb = 0; kb < K; kb += 32) {
        v16h a[MT];
#pragma unroll
        for (int mi = 0; mi < MT; mi++) {
            v8h lo = *(const v8h*)(arow[mi] + kb);
            v8h hi = *(const v8h*)(arow[mi] + kb + 16);
            a[mi] = __builtin_shufflevector(lo, hi, 0, 1, 2, 3, 4, 5, 6, 7,
                                            8, 9, 10, 11, 12, 13, 14, 15);
        }
        v16h bf[NT];
#pragma unroll
        for (int j = 0; j < NT; j++) {
            v8h lo = *(const v8h*)(brow[j] + kb);
            v8h hi = *(const v8h*)(brow[j] + kb + 16);
            bf[j] = __builtin_shufflevector(lo, hi, 0, 1, 2, 3, 4, 5, 6, 7,
                                            8, 9, 10, 11, 12, 13, 14, 15);
        }
#pragma unroll
        for (int j = 0; j < NT; j++)
#pragma unroll
            for (int mi = 0; mi < MT; mi++)
                acc[mi][j] = __builtin_amdgcn_wmma_f32_16x16x32_f16(
                    false, a[mi], false, bf[j], (short)0, acc[mi][j],
                    false, false);
    }
#pragma unroll
    for (int mi = 0; mi < MT; mi++) {
#pragma unroll
        for (int j = 0; j < NT; j++) {
            int n0 = tng * NT * 16 + j * 16;
#pragma unroll
            for (int rr = 0; rr < 8; rr++) {
                int mrow = tmg * MT * 16 + mi * 16 + rr + 8 * g;
                Out[(size_t)mrow * OUT_STRIDE + n0 + r] = acc[mi][j][rr];
            }
        }
    }
}

// ---------------------------------------------------------------------------
// Depthwise causal conv(4) + bias + SiLU; input = x1 half of xz (stride 1024)
// ---------------------------------------------------------------------------
__global__ void conv_silu_kernel(const float* __restrict__ xz,
                                 const float* __restrict__ cw,   // 512 x 4
                                 const float* __restrict__ cb,   // 512
                                 float* __restrict__ x1c,
                                 _Float16* __restrict__ x1c16) {
    int t = blockIdx.x * blockDim.x + threadIdx.x;
    if (t >= MM * D_INNER) return;
    int m = t >> 9;            // t / 512
    int e = t & (D_INNER - 1); // t % 512
    int l = m & (LL - 1);      // position within sequence
    const float* w = cw + e * 4;
    float acc = cb[e];
#pragma unroll
    for (int j = 0; j < 4; j++) {
        int li = l - 3 + j;
        if (li >= 0) acc += w[j] * xz[(size_t)(m - 3 + j) * 1024 + e];
    }
    float sv = acc / (1.0f + __expf(-acc));   // SiLU
    x1c[t]   = sv;
    x1c16[t] = (_Float16)sv;
}

// ---------------------------------------------------------------------------
// Selective scan: grid (B, D_INNER/128), 128 threads = 1 channel each.
// Per-token x_dbl row (dt_r[16] | B[16] | C[16]) staged via LDS in chunks of 8.
// dt = softplus(dt_r . W_dt[e] + b_dt[e]); h[s] = exp(dt*A[s])h + dt*x*B[s];
// y = h.C + D*x; y *= silu(z); write f16 for the output GEMM.
// ---------------------------------------------------------------------------
__global__ void scan_kernel(const float* __restrict__ xdbl,  // MM x 48
                            const float* __restrict__ x1c,   // MM x 512
                            const float* __restrict__ xz,    // MM x 1024 (z at +512)
                            const float* __restrict__ Wdt,   // 512 x 16
                            const float* __restrict__ bdt,   // 512
                            const float* __restrict__ Alog,  // 512 x 16
                            const float* __restrict__ Dp,    // 512
                            _Float16* __restrict__ y16) {    // MM x 512
    const int CH = 8;
    __shared__ float lds[CH * N_DBL];   // 8 tokens x 48
    int b = blockIdx.x;
    int e = blockIdx.y * blockDim.x + threadIdx.x;
    float wdt[DT_RANK], A[D_STATE], h[D_STATE];
#pragma unroll
    for (int i = 0; i < 16; i++) {
        wdt[i] = Wdt[e * 16 + i];
        A[i]   = -__expf(Alog[e * 16 + i]);
        h[i]   = 0.0f;
    }
    float bd = bdt[e], Dv = Dp[e];
    size_t m0 = (size_t)b * LL;
    for (int c0 = 0; c0 < LL; c0 += CH) {
        __syncthreads();   // protect previous chunk's readers
        for (int i = threadIdx.x; i < CH * N_DBL; i += blockDim.x)
            lds[i] = xdbl[(m0 + c0) * N_DBL + i];   // 8 contiguous rows
        __syncthreads();
#pragma unroll
        for (int c = 0; c < CH; c++) {
            size_t m = m0 + c0 + c;
            const float* row = lds + c * N_DBL;
            float dt = bd;
#pragma unroll
            for (int i = 0; i < 16; i++) dt += wdt[i] * row[i];
            dt = (dt > 20.0f) ? dt : log1pf(__expf(dt));   // softplus
            float xt  = x1c[m * D_INNER + e];
            float dtx = dt * xt;
            float y = 0.0f;
#pragma unroll
            for (int s = 0; s < 16; s++) {
                h[s] = __expf(dt * A[s]) * h[s] + dtx * row[16 + s];
                y += h[s] * row[32 + s];
            }
            y += Dv * xt;
            float z = xz[m * 1024 + 512 + e];
            y *= z / (1.0f + __expf(-z));                  // * silu(z)
            y16[m * D_INNER + e] = (_Float16)y;
        }
    }
}

// ---------------------------------------------------------------------------
// out[b,l,v,d] = h_v + mean_v(h)   (d_out already holds h_v from GEMM3)
// ---------------------------------------------------------------------------
__global__ void combine_kernel(float* __restrict__ out) {
    int t = blockIdx.x * blockDim.x + threadIdx.x;
    if (t >= MM * D_MODEL) return;
    int m = t >> 8;            // token
    int d = t & 255;
    float* p = out + (size_t)m * N_VIEWS * D_MODEL + d;
    float a0 = p[0], a1 = p[256], a2 = p[512], a3 = p[768];
    float mean = 0.25f * (a0 + a1 + a2 + a3);
    p[0] = a0 + mean; p[256] = a1 + mean; p[512] = a2 + mean; p[768] = a3 + mean;
}

// ---------------------------------------------------------------------------
extern "C" void kernel_launch(void* const* d_in, const int* in_sizes, int n_in,
                              void* d_out, int out_size, void* d_ws, size_t ws_size,
                              hipStream_t stream) {
    (void)in_sizes; (void)n_in; (void)out_size; (void)ws_size;
    const float* x      = (const float*)d_in[0];
    const float* ln_g   = (const float*)d_in[1];
    const float* ln_b   = (const float*)d_in[2];
    const float* W_in   = (const float*)d_in[3];
    const float* conv_w = (const float*)d_in[4];
    const float* conv_b = (const float*)d_in[5];
    const float* W_x    = (const float*)d_in[6];
    const float* W_dt   = (const float*)d_in[7];
    const float* b_dt   = (const float*)d_in[8];
    const float* A_log  = (const float*)d_in[9];
    const float* Dp     = (const float*)d_in[10];
    const float* W_out  = (const float*)d_in[11];
    float* out = (float*)d_out;

    // workspace sub-allocation (256B aligned)
    char* ws = (char*)d_ws;
    size_t off = 0;
    auto alloc = [&](size_t bytes) -> char* {
        char* p = ws + off;
        off += (bytes + 255) & ~(size_t)255;
        return p;
    };
    _Float16* Win16  = (_Float16*)alloc((size_t)N_VIEWS * 1024 * 256 * 2);
    _Float16* Wx16   = (_Float16*)alloc((size_t)N_VIEWS * N_DBL * 512 * 2);
    _Float16* Wout16 = (_Float16*)alloc((size_t)N_VIEWS * 256 * 512 * 2);
    _Float16* xn16   = (_Float16*)alloc((size_t)MM * 256 * 2);
    float*    xzbuf  = (float*)   alloc((size_t)MM * 1024 * 4);
    float*    x1c    = (float*)   alloc((size_t)MM * 512 * 4);
    _Float16* x1c16  = (_Float16*)alloc((size_t)MM * 512 * 2);
    float*    xdbl   = (float*)   alloc((size_t)MM * N_DBL * 4);
    _Float16* y16    = (_Float16*)alloc((size_t)MM * 512 * 2);

    // weight conversion (deterministic; redone each call)
    {
        int n1 = N_VIEWS * 1024 * 256;
        int n2 = N_VIEWS * N_DBL * 512;
        int n3 = N_VIEWS * 256 * 512;
        f32_to_f16_kernel<<<(n1 + 255) / 256, 256, 0, stream>>>(W_in,  Win16,  n1);
        f32_to_f16_kernel<<<(n2 + 255) / 256, 256, 0, stream>>>(W_x,   Wx16,   n2);
        f32_to_f16_kernel<<<(n3 + 255) / 256, 256, 0, stream>>>(W_out, Wout16, n3);
    }

    // blocks for MT=2 GEMMs: waves = (M/32) * n_groups, 8 waves per block
    auto gemm_blocks = [](int M, int n_grp) {
        int waves = (M / 32) * n_grp;
        return (waves * 32 + 255) / 256;
    };

    for (int v = 0; v < N_VIEWS; v++) {
        // 1) LayerNorm -> f16
        layernorm_f16_kernel<<<MM / 8, 256, 0, stream>>>(
            x, ln_g + v * 256, ln_b + v * 256, xn16, v);

        // 2) xz = xn @ W_in^T   (16384 x 1024, K=256): 16 n-groups of 64
        wmma_gemm_f16_kernel<1024, 256, 4, 2, 1024>
            <<<gemm_blocks(MM, 16), 256, 0, stream>>>(
                xn16, Win16 + (size_t)v * 1024 * 256, xzbuf, MM);

        // 3) depthwise conv + SiLU
        conv_silu_kernel<<<(MM * D_INNER) / 256, 256, 0, stream>>>(
            xzbuf, conv_w + v * 512 * 4, conv_b + v * 512, x1c, x1c16);

        // 4) x_dbl = x1 @ W_x^T  (16384 x 48, K=512): 1 n-group of 48
        wmma_gemm_f16_kernel<N_DBL, 512, 3, 2, N_DBL>
            <<<gemm_blocks(MM, 1), 256, 0, stream>>>(
                x1c16, Wx16 + (size_t)v * N_DBL * 512, xdbl, MM);

        // 5) selective scan + gating -> y16
        scan_kernel<<<dim3(BB, D_INNER / 128), 128, 0, stream>>>(
            xdbl, x1c, xzbuf,
            W_dt + (size_t)v * 512 * 16, b_dt + v * 512,
            A_log + (size_t)v * 512 * 16, Dp + v * 512, y16);

        // 6) h_v = y @ W_out^T -> d_out[:, :, v, :]  (row stride V*256)
        wmma_gemm_f16_kernel<256, 512, 4, 2, N_VIEWS * 256>
            <<<gemm_blocks(MM, 4), 256, 0, stream>>>(
                y16, Wout16 + (size_t)v * 256 * 512, out + v * 256, MM);
    }

    // 7) out = h + mean over views
    combine_kernel<<<(MM * D_MODEL) / 256, 256, 0, stream>>>(out);
}
